// TropicalAttention_52003464020141
// MI455X (gfx1250) — matrix operand order repaired and checked by reading, not compiled, not run on hardware
//
#include <hip/hip_runtime.h>
#include <hip/hip_bf16.h>

typedef _Float16 v16h __attribute__((ext_vector_type(16)));
typedef _Float16 v8h  __attribute__((ext_vector_type(8)));
typedef _Float16 v4h  __attribute__((ext_vector_type(4)));
typedef _Float16 v2h  __attribute__((ext_vector_type(2)));
typedef float    v8f  __attribute__((ext_vector_type(8)));

#define NEG_INF (-__builtin_inff())

// Problem constants (B=4, T=2048, d=512, H=8, dh=64)
#define BT   8192     // B*T
#define DD   512      // d
#define HH   8
#define DH   64
#define TT   2048
#define BHT  65536    // B*H*T

__device__ __forceinline__ v8f wmma16(v16h a, v16h b, v8f c) {
  return __builtin_amdgcn_wmma_f32_16x16x32_f16(false, a, false, b, (short)0, c, false, false);
}

// A-operand fragment, 16x32 f16 (row-major source, stride in halves).
// Lane L<16: row M=L, K = {0..7, 16..23}; lane L>=16: row M=L-16, K = {8..15, 24..31}.
__device__ __forceinline__ v16h load_fragA(const _Float16* __restrict__ base, int stride, int lane) {
  const int r  = lane & 15;
  const int hi = (lane >> 4) & 1;
  const _Float16* p = base + r * stride + hi * 8;
  v8h lo = *(const v8h*)(p);
  v8h hh = *(const v8h*)(p + 16);
  return __builtin_shufflevector(lo, hh, 0,1,2,3,4,5,6,7,8,9,10,11,12,13,14,15);
}

// B-operand fragment, 32x16 f16 where B(k,n) = src[n*stride + k] (i.e. src rows = output cols).
// Lane L<16: col N=L, K = {0..7, 16..23}; lane L>=16: col N=L-16, K = {8..15, 24..31}.
__device__ __forceinline__ v16h load_fragB(const _Float16* __restrict__ base, int stride, int lane) {
  const int n  = lane & 15;
  const int hi = (lane >> 4) & 1;
  const _Float16* p = base + n * stride + hi * 8;
  v8h lo = *(const v8h*)(p);
  v8h hh = *(const v8h*)(p + 16);
  return __builtin_shufflevector(lo, hh, 0,1,2,3,4,5,6,7,8,9,10,11,12,13,14,15);
}

// ---------------------------------------------------------------- K0: f32 -> f16
__global__ void __launch_bounds__(256) cvt_f32_f16(const float* __restrict__ src,
                                                   _Float16* __restrict__ dst, int n4) {
  int i = blockIdx.x * blockDim.x + threadIdx.x;
  if (i < n4) {
    float4 f = ((const float4*)src)[i];
    v4h o;
    o[0] = (_Float16)f.x; o[1] = (_Float16)f.y; o[2] = (_Float16)f.z; o[3] = (_Float16)f.w;
    ((v4h*)dst)[i] = o;
  }
}

// ---------------------------------------------------------------- K1: QKV projection (+ row max, exp)
// Tile per wave: 16 rows x 64 cols (one head slice).
// Tile order: ((w*8 + h) * 512 + mt) so the 4 waves of a WG share (w,h) -> identical
// B-fragment addresses (WGP$-coalesced), differing only in the row tile.
// Software-pipelined k-loop: next slab's A/B fragments are loaded while the
// current slab's WMMAs execute (forces two live register sets -> partial waits).
__global__ void __launch_bounds__(128) proj_qkv(
    const _Float16* __restrict__ xh,
    const _Float16* __restrict__ Wqh, const _Float16* __restrict__ Wkh, const _Float16* __restrict__ Wvh,
    _Float16* __restrict__ P, _Float16* __restrict__ Ke, _Float16* __restrict__ Vh,
    float* __restrict__ mQ, float* __restrict__ mK) {
  const int lane = threadIdx.x & 31;
  const int wv   = threadIdx.x >> 5;
  const int tile = blockIdx.x * 4 + wv;            // 12288 tiles
  const int mt   = tile & 511;                     // row tile (varies across waves)
  const int wh   = tile >> 9;                      // 0..23 (shared by the 4 waves)
  const int w    = wh >> 3;                        // 0=Q 1=K 2=V
  const int h    = wh & 7;
  const int mrow0 = mt * 16;
  const int colbase = h * DH;
  const _Float16* W = (w == 0) ? Wqh : (w == 1) ? Wkh : Wvh;
  const _Float16* Arow = xh + mrow0 * DD;

  v8f acc[4];
  const v8f zv = {0.f,0.f,0.f,0.f,0.f,0.f,0.f,0.f};
#pragma unroll
  for (int t = 0; t < 4; ++t) acc[t] = zv;

  // prologue: load slab 0
  v16h a_c = load_fragA(Arow, DD, lane);
  v16h b_c[4];
#pragma unroll
  for (int t = 0; t < 4; ++t)
    b_c[t] = load_fragB(W + (colbase + t * 16) * DD, DD, lane);

#pragma unroll 2
  for (int kc = 32; kc < DD; kc += 32) {
    __builtin_prefetch(Arow + (lane & 15) * DD + kc + 32, 0, 1);
    // issue next slab's loads first
    v16h a_n = load_fragA(Arow + kc, DD, lane);
    v16h b_n[4];
#pragma unroll
    for (int t = 0; t < 4; ++t)
      b_n[t] = load_fragB(W + (colbase + t * 16) * DD + kc, DD, lane);
    // compute on current slab (its loads landed while previous WMMAs ran)
#pragma unroll
    for (int t = 0; t < 4; ++t)
      acc[t] = wmma16(a_c, b_c[t], acc[t]);
    a_c = a_n;
#pragma unroll
    for (int t = 0; t < 4; ++t) b_c[t] = b_n[t];
  }
#pragma unroll
  for (int t = 0; t < 4; ++t)
    acc[t] = wmma16(a_c, b_c[t], acc[t]);

  const int bb   = mrow0 >> 11;        // batch
  const int trow = mrow0 & (TT - 1);
  const int bh   = bb * HH + h;
  const int prow0 = bh * TT + trow;
  const int ln  = lane & 15;
  const int rr0 = (lane < 16) ? 0 : 8;

  if (w == 2) {
#pragma unroll
    for (int r = 0; r < 8; ++r) {
      const int row = prow0 + rr0 + r;
#pragma unroll
      for (int t = 0; t < 4; ++t)
        Vh[row * DH + t * 16 + ln] = (_Float16)acc[t][r];
    }
  } else {
    _Float16* O = (w == 0) ? P : Ke;
    float*    M = (w == 0) ? mQ : mK;
#pragma unroll
    for (int r = 0; r < 8; ++r) {
      float m = fmaxf(fmaxf(acc[0][r], acc[1][r]), fmaxf(acc[2][r], acc[3][r]));
#pragma unroll
      for (int off = 1; off < 16; off <<= 1)   // reduce within 16-lane half
        m = fmaxf(m, __shfl_xor(m, off, 32));
      const int row = prow0 + rr0 + r;
#pragma unroll
      for (int t = 0; t < 4; ++t)
        O[row * DH + t * 16 + ln] = (_Float16)__expf(acc[t][r] - m);
      if (ln == 0) M[row] = m;
    }
  }
}

__device__ __forceinline__ void topk_insert(float s, int j, float (&t)[8], int (&ji)[8]) {
  if (s <= t[0]) return;       // below current 8th-largest
  t[0] = s; ji[0] = j;
#pragma unroll
  for (int k = 0; k < 7; ++k) {
    if (t[k] > t[k + 1]) {
      float a = t[k]; t[k] = t[k + 1]; t[k + 1] = a;
      int   b = ji[k]; ji[k] = ji[k + 1]; ji[k + 1] = b;
    }
  }
}

// ---------------------------------------------------------------- K2: scores (WMMA) + streaming top-8 + softmax + ctx gather
// Block: 4 waves, each wave owns 16 query rows of one (b,h); waves share bh so
// their Ke B-fragment loads are identical for overlapping key tiles.
// Ke fragments are pipelined across the key loop: tile jt+1's loads are issued
// before tile jt's WMMA/scan phase, hiding them behind the LDS scan + barriers.
__global__ void __launch_bounds__(128) attn_topk(
    const _Float16* __restrict__ P, const _Float16* __restrict__ Ke, const _Float16* __restrict__ Vh,
    const float* __restrict__ mQ, const float* __restrict__ mK,
    _Float16* __restrict__ ctxh, const int* __restrict__ causal_p) {
  __shared__ float smem[4][16][17];
  __shared__ float mks[4][16];
  __shared__ float wtab[4][16][8];
  __shared__ int   jtab[4][16][8];

  const int lane = threadIdx.x & 31;
  const int wv   = threadIdx.x >> 5;
  const int bh   = blockIdx.x >> 5;
  const int qt   = blockIdx.x & 31;
  const int causal = causal_p[0];
  const int qbase = qt * 64;
  const int qrow0 = qbase + wv * 16;
  const int rbase = bh * TT;

  // Q fragments (held for the whole key sweep): exp(Q - mQ), 16 rows x 64ch
  const v16h a0 = load_fragA(P + (rbase + qrow0) * DH + 0,  DH, lane);
  const v16h a1 = load_fragA(P + (rbase + qrow0) * DH + 32, DH, lane);

  float mql = 0.f;
  if (lane < 16) mql = mQ[rbase + qrow0 + lane];

  float tv[8]; int tj[8];
#pragma unroll
  for (int k = 0; k < 8; ++k) { tv[k] = NEG_INF; tj[k] = 0; }

  const int ntiles = causal ? (qbase + 64) / 16 : (TT / 16);   // WG-uniform

  // prologue: key tile 0 is active for every wave
  v16h b0c = load_fragB(Ke + rbase * DH + 0,  DH, lane);
  v16h b1c = load_fragB(Ke + rbase * DH + 32, DH, lane);

  for (int jt = 0; jt < ntiles; ++jt) {
    const int j0 = jt * 16;
    const bool active = (!causal) || (j0 <= qrow0 + 15);                         // wave-uniform
    const bool pre = ((jt + 1) < ntiles) &&
                     ((!causal) || (j0 + 16 <= qrow0 + 15));                     // wave-uniform
    v16h b0n, b1n;
    if (pre) {   // issue next tile's loads; they complete during scan+barriers
      b0n = load_fragB(Ke + (rbase + j0 + 16) * DH + 0,  DH, lane);
      b1n = load_fragB(Ke + (rbase + j0 + 16) * DH + 32, DH, lane);
    }
    if (active) {
      v8f acc = {0.f,0.f,0.f,0.f,0.f,0.f,0.f,0.f};
      acc = wmma16(a0, b0c, acc);
      acc = wmma16(a1, b1c, acc);
      const int ln  = lane & 15;
      const int rr0 = (lane < 16) ? 0 : 8;
#pragma unroll
      for (int r = 0; r < 8; ++r) smem[wv][rr0 + r][ln] = acc[r];
      if (lane < 16) mks[wv][lane] = mK[rbase + j0 + lane];
    }
    __syncthreads();
    if (active && lane < 16) {
      const int row = qrow0 + lane;
#pragma unroll 4
      for (int c = 0; c < 16; ++c) {
        const int j = j0 + c;
        if (!causal || (j <= row)) {
          float s = mql + mks[wv][c] + __logf(smem[wv][lane][c]);
          topk_insert(s, j, tv, tj);
        }
      }
    }
    __syncthreads();
    if (pre) { b0c = b0n; b1c = b1n; }
  }

  // in-register softmax over the kept top-8 (tv[7] is the row max)
  if (lane < 16) {
    const float mx = tv[7];
    float e[8]; float z = 0.f;
#pragma unroll
    for (int k = 0; k < 8; ++k) { e[k] = __expf(tv[k] - mx); z += e[k]; }
    const float inv = 1.f / z;
#pragma unroll
    for (int k = 0; k < 8; ++k) { wtab[wv][lane][k] = e[k] * inv; jtab[wv][lane][k] = tj[k]; }
  }
  __syncthreads();

  // ctx = sum_k w_k * V[j_k, :]; each lane handles cols (2*lane, 2*lane+1)
  const int bb = bh >> 3, h = bh & 7;
  for (int r = 0; r < 16; ++r) {
    const int trow = qrow0 + r;
    float c0 = 0.f, c1 = 0.f;
#pragma unroll
    for (int k = 0; k < 8; ++k) {
      const float wgt = wtab[wv][r][k];
      const int   j   = jtab[wv][r][k];
      v2h v = *(const v2h*)(Vh + (rbase + j) * DH + 2 * lane);
      c0 += wgt * (float)v.x;
      c1 += wgt * (float)v.y;
    }
    v2h o; o.x = (_Float16)c0; o.y = (_Float16)c1;
    *(v2h*)(ctxh + (bb * TT + trow) * DD + h * DH + 2 * lane) = o;
  }
}

// ---------------------------------------------------------------- K3: output projection -> f32
// Tile order: (ct * 512 + mt) so the 4 waves of a WG share the Wo column block.
__global__ void __launch_bounds__(128) proj_out(
    const _Float16* __restrict__ ctxh, const _Float16* __restrict__ Woh,
    float* __restrict__ out) {
  const int lane = threadIdx.x & 31;
  const int wv   = threadIdx.x >> 5;
  const int tile = blockIdx.x * 4 + wv;      // 4096 tiles: 8 coltiles(64) x 512 rowtiles
  const int mt   = tile & 511;
  const int ct   = tile >> 9;
  const int mrow0 = mt * 16;
  const int colbase = ct * 64;
  const _Float16* Arow = ctxh + mrow0 * DD;

  v8f acc[4];
  const v8f zv = {0.f,0.f,0.f,0.f,0.f,0.f,0.f,0.f};
#pragma unroll
  for (int t = 0; t < 4; ++t) acc[t] = zv;

  v16h a_c = load_fragA(Arow, DD, lane);
  v16h b_c[4];
#pragma unroll
  for (int t = 0; t < 4; ++t)
    b_c[t] = load_fragB(Woh + (colbase + t * 16) * DD, DD, lane);

#pragma unroll 2
  for (int kc = 32; kc < DD; kc += 32) {
    __builtin_prefetch(Arow + (lane & 15) * DD + kc + 32, 0, 1);
    v16h a_n = load_fragA(Arow + kc, DD, lane);
    v16h b_n[4];
#pragma unroll
    for (int t = 0; t < 4; ++t)
      b_n[t] = load_fragB(Woh + (colbase + t * 16) * DD + kc, DD, lane);
#pragma unroll
    for (int t = 0; t < 4; ++t)
      acc[t] = wmma16(a_c, b_c[t], acc[t]);
    a_c = a_n;
#pragma unroll
    for (int t = 0; t < 4; ++t) b_c[t] = b_n[t];
  }
#pragma unroll
  for (int t = 0; t < 4; ++t)
    acc[t] = wmma16(a_c, b_c[t], acc[t]);

  const int ln  = lane & 15;
  const int rr0 = (lane < 16) ? 0 : 8;
#pragma unroll
  for (int r = 0; r < 8; ++r) {
    const int row = mrow0 + rr0 + r;
#pragma unroll
    for (int t = 0; t < 4; ++t)
      out[row * DD + colbase + t * 16 + ln] = acc[t][r];
  }
}

// ----------------------------------------------------------------
extern "C" void kernel_launch(void* const* d_in, const int* in_sizes, int n_in,
                              void* d_out, int out_size, void* d_ws, size_t ws_size,
                              hipStream_t stream) {
  (void)in_sizes; (void)n_in; (void)out_size; (void)ws_size;
  const float* x  = (const float*)d_in[0];
  const float* Wq = (const float*)d_in[1];
  const float* Wk = (const float*)d_in[2];
  const float* Wv = (const float*)d_in[3];
  const float* Wo = (const float*)d_in[4];
  const int* causal = (const int*)d_in[5];
  float* out = (float*)d_out;

  size_t off = 0;
  char* ws = (char*)d_ws;
  auto take = [&](size_t bytes) -> char* {
    char* p = ws + off;
    off += (bytes + 255) & ~(size_t)255;
    return p;
  };
  _Float16* xh   = (_Float16*)take((size_t)BT * DD * 2);
  _Float16* Wqh  = (_Float16*)take((size_t)DD * DD * 2);
  _Float16* Wkh  = (_Float16*)take((size_t)DD * DD * 2);
  _Float16* Wvh  = (_Float16*)take((size_t)DD * DD * 2);
  _Float16* Woh  = (_Float16*)take((size_t)DD * DD * 2);
  _Float16* P    = (_Float16*)take((size_t)BHT * DH * 2);
  _Float16* Ke   = (_Float16*)take((size_t)BHT * DH * 2);
  _Float16* Vh   = (_Float16*)take((size_t)BHT * DH * 2);
  float*    mQ   = (float*)take((size_t)BHT * 4);
  float*    mK   = (float*)take((size_t)BHT * 4);
  _Float16* ctxh = (_Float16*)take((size_t)BT * DD * 2);

  // K0: f32 -> f16 conversions
  {
    int n4x = BT * DD / 4;        // 1,048,576
    int n4w = DD * DD / 4;        //    65,536
    cvt_f32_f16<<<(n4x + 255) / 256, 256, 0, stream>>>(x,  xh,  n4x);
    cvt_f32_f16<<<(n4w + 255) / 256, 256, 0, stream>>>(Wq, Wqh, n4w);
    cvt_f32_f16<<<(n4w + 255) / 256, 256, 0, stream>>>(Wk, Wkh, n4w);
    cvt_f32_f16<<<(n4w + 255) / 256, 256, 0, stream>>>(Wv, Wvh, n4w);
    cvt_f32_f16<<<(n4w + 255) / 256, 256, 0, stream>>>(Wo, Woh, n4w);
  }

  // K1: Q/K/V projections + stable-exp transform (12288 wave tiles / 4 per block)
  proj_qkv<<<3072, 128, 0, stream>>>(xh, Wqh, Wkh, Wvh, P, Ke, Vh, mQ, mK);

  // K2: tropical scores + top-8 + softmax + ctx (32 bh * 32 qtiles)
  attn_topk<<<1024, 128, 0, stream>>>(P, Ke, Vh, mQ, mK, ctxh, causal);

  // K3: output projection (4096 wave tiles / 4 per block)
  proj_out<<<1024, 128, 0, stream>>>(ctxh, Woh, out);
}
